// Multi_modal_Joint_Attention_48850958024955
// MI455X (gfx1250) — matrix operand order repaired and checked by reading, not compiled, not run on hardware
//
#include <hip/hip_runtime.h>
#include <hip/hip_bf16.h>

// ---------------------------------------------------------------------------
// Multi-modal joint attention, CDNA5 (gfx1250, wave32) WMMA implementation.
// All GEMM operands are f16 end-to-end; accumulation in f32 via
// v_wmma_f32_16x16x32_f16. fp32 is kept for scores (pre-softmax), gating
// linear outputs, and final elementwise math.
// ---------------------------------------------------------------------------

typedef __attribute__((ext_vector_type(16))) _Float16 v16h;
typedef __attribute__((ext_vector_type(8)))  _Float16 v8h;
typedef __attribute__((ext_vector_type(8)))  float    v8f;

#define WMMA_F16(a, b, c) \
  __builtin_amdgcn_wmma_f32_16x16x32_f16(false, (a), false, (b), (short)0, (c), false, false)

// ---- Fragment loaders (layouts per CDNA5 ISA 7.12.2) -----------------------
// A 16x32 f16 frag from a row-major f16 [M x lda] matrix.
// Lane L (L&15 = row), h = L>>4:  halves[0..7]  = K (h*8    .. h*8+7)
//                                 halves[8..15] = K (16+h*8 .. 16+h*8+7)
__device__ __forceinline__ v16h load_a_f16(const _Float16* A, int lda, int row0, int k0, int lane) {
  int r = row0 + (lane & 15);
  int h = lane >> 4;
  const _Float16* p = A + (size_t)r * lda + k0 + h * 8;
  v8h lo = *(const v8h*)(p);        // one b128
  v8h hi = *(const v8h*)(p + 16);   // one b128
  v16h a;
#pragma unroll
  for (int i = 0; i < 8; ++i) { a[i] = lo[i]; a[8 + i] = hi[i]; }
  return a;
}

// B 32x16 f16 frag from Bt[N][K] (ldb = K): lane col = L&15, h = L>>4,
// halves[0..7] = K (h*8 .. h*8+7), halves[8..15] = K (16+h*8 .. 16+h*8+7) -> two b128 loads.
__device__ __forceinline__ v16h load_b_f16(const _Float16* Bt, int ldb, int col0, int k0, int lane) {
  int n = col0 + (lane & 15);
  int h = lane >> 4;
  const _Float16* p = Bt + (size_t)n * ldb + k0 + h * 8;
  v8h lo = *(const v8h*)(p);
  v8h hi = *(const v8h*)(p + 16);
  v16h b;
#pragma unroll
  for (int i = 0; i < 8; ++i) { b[i] = lo[i]; b[8 + i] = hi[i]; }
  return b;
}

// Implicit A frag for the bilinear: P[s, i*256+j] = vs * a16[s, j]
// (v_pk_mul_f16 on loaded halves).
__device__ __forceinline__ v16h make_p_frag16(const _Float16* arow, int j0, int h, _Float16 vs) {
  const _Float16* p = arow + j0 + h * 8;
  v8h lo = *(const v8h*)(p);
  v8h hi = *(const v8h*)(p + 16);
  v16h a;
#pragma unroll
  for (int i = 0; i < 8; ++i) { a[i] = lo[i] * vs; a[8 + i] = hi[i] * vs; }
  return a;
}

// ---- Generic WMMA GEMM: C = A (f16,[M,K]) @ Bt (f16,[N][K]) + bias ----------
// Outputs are optional: C (f32) and/or C16 (f16), both row-major [M,N].
// Wave computes 32x32 (2x2 frags); block = 4 waves (2x2) -> 64x64 tile.
// grid = (N/64, M/64, batch); per-batch element strides sA, sB, sC.
__global__ void __launch_bounds__(128)
gemm_wmma_f16(const _Float16* __restrict__ A, const _Float16* __restrict__ Bt,
              const float* __restrict__ bias, float* __restrict__ C,
              _Float16* __restrict__ C16, int M, int N, int K,
              long sA, long sB, long sC) {
  int bz = blockIdx.z;
  A  += (size_t)bz * sA;
  Bt += (size_t)bz * sB;
  if (C)   C   += (size_t)bz * sC;
  if (C16) C16 += (size_t)bz * sC;
  int lane = threadIdx.x & 31;
  int wave = threadIdx.x >> 5;
  int row0 = blockIdx.y * 64 + (wave >> 1) * 32;
  int col0 = blockIdx.x * 64 + (wave & 1) * 32;

  v8f acc[2][2] = {};
  for (int k0 = 0; k0 < K; k0 += 32) {
    v16h a0 = load_a_f16(A, K, row0,      k0, lane);
    v16h a1 = load_a_f16(A, K, row0 + 16, k0, lane);
    v16h b0 = load_b_f16(Bt, K, col0,      k0, lane);
    v16h b1 = load_b_f16(Bt, K, col0 + 16, k0, lane);
    acc[0][0] = WMMA_F16(a0, b0, acc[0][0]);
    acc[0][1] = WMMA_F16(a0, b1, acc[0][1]);
    acc[1][0] = WMMA_F16(a1, b0, acc[1][0]);
    acc[1][1] = WMMA_F16(a1, b1, acc[1][1]);
  }

  int h  = lane >> 4;
  int cn = lane & 15;
#pragma unroll
  for (int mi = 0; mi < 2; ++mi)
#pragma unroll
    for (int ni = 0; ni < 2; ++ni) {
      int c  = col0 + ni * 16 + cn;
      float bv = bias ? bias[c] : 0.0f;
#pragma unroll
      for (int v = 0; v < 8; ++v) {
        int r = row0 + mi * 16 + v + h * 8;
        float val = acc[mi][ni][v] + bv;
        if (C)   C[(size_t)r * N + c]   = val;
        if (C16) C16[(size_t)r * N + c] = (_Float16)val;
      }
    }
}

// ---- Bilinear: out[s,k] = sum_ij cfv[s,i] * bil16[k][i*256+j] * cfa[s,j] ---
// Implicit-A GEMM over KK = i*256+j (65536). bil16 is [N=256][KK] f16.
// Wave computes 64x32 (4 M-frags x 2 N-frags = 8 WMMAs / K-step).
// block = 4 waves (2x2) -> 128x64 tile; grid = (256/64, 2048/128).
__global__ void __launch_bounds__(128)
bilinear_wmma(const _Float16* __restrict__ cfv, const _Float16* __restrict__ cfa,
              const _Float16* __restrict__ bil16, float* __restrict__ out, int Dm) {
  int lane = threadIdx.x & 31;
  int wave = threadIdx.x >> 5;
  int row0 = blockIdx.y * 128 + (wave >> 1) * 64;
  int col0 = blockIdx.x * 64 + (wave & 1) * 32;
  int h = lane >> 4;
  const int KKtot = Dm * Dm;

  int r[4];
  const _Float16* arow[4];
#pragma unroll
  for (int mi = 0; mi < 4; ++mi) {
    r[mi] = row0 + mi * 16 + (lane & 15);
    arow[mi] = cfa + (size_t)r[mi] * Dm;
  }

  v8f acc[4][2] = {};
  for (int i = 0; i < Dm; ++i) {
    _Float16 vs[4];
#pragma unroll
    for (int mi = 0; mi < 4; ++mi) vs[mi] = cfv[(size_t)r[mi] * Dm + i];
#pragma unroll
    for (int j0 = 0; j0 < 256; j0 += 32) {
      int KK = i * Dm + j0;
      v16h b0 = load_b_f16(bil16, KKtot, col0,      KK, lane);
      v16h b1 = load_b_f16(bil16, KKtot, col0 + 16, KK, lane);
      // Prefetch the next K-slab of the streaming (L2-resident) B operand.
      __builtin_prefetch(bil16 + (size_t)(col0 + (lane & 15)) * KKtot + KK + 32, 0, 1);
#pragma unroll
      for (int mi = 0; mi < 4; ++mi) {
        v16h a = make_p_frag16(arow[mi], j0, h, vs[mi]);
        acc[mi][0] = WMMA_F16(a, b0, acc[mi][0]);
        acc[mi][1] = WMMA_F16(a, b1, acc[mi][1]);
      }
    }
  }

  int cn = lane & 15;
#pragma unroll
  for (int mi = 0; mi < 4; ++mi)
#pragma unroll
    for (int ni = 0; ni < 2; ++ni) {
      int c = col0 + ni * 16 + cn;
#pragma unroll
      for (int v = 0; v < 8; ++v) {
        int rr = row0 + mi * 16 + v + h * 8;
        out[(size_t)rr * Dm + c] = acc[mi][ni][v];
      }
    }
}

// ---- Row softmax over T=512, then * norm_fact; f32 in, f16 out -------------
__global__ void __launch_bounds__(256)
softmax_scale_f16(const float* __restrict__ X, _Float16* __restrict__ Y, int T, float scale) {
  int row  = blockIdx.x * 8 + (threadIdx.x >> 5);
  int lane = threadIdx.x & 31;
  const float* p = X + (size_t)row * T;
  _Float16*    q = Y + (size_t)row * T;
  float v[16];
  float mx = -3.4e38f;
#pragma unroll
  for (int i = 0; i < 16; ++i) { v[i] = p[lane + i * 32]; mx = fmaxf(mx, v[i]); }
#pragma unroll
  for (int off = 16; off >= 1; off >>= 1) mx = fmaxf(mx, __shfl_xor(mx, off, 32));
  float s = 0.f;
#pragma unroll
  for (int i = 0; i < 16; ++i) { v[i] = __expf(v[i] - mx); s += v[i]; }
#pragma unroll
  for (int off = 16; off >= 1; off >>= 1) s += __shfl_xor(s, off, 32);
  float rs = scale / s;
#pragma unroll
  for (int i = 0; i < 16; ++i) q[lane + i * 32] = (_Float16)(v[i] * rs);
}

// ---- Small elementwise / layout kernels ------------------------------------
__global__ void cast_f16_kernel(const float* __restrict__ in, _Float16* __restrict__ out, long n) {
  long i = (long)blockIdx.x * blockDim.x + threadIdx.x;
  if (i < n) out[i] = (_Float16)in[i];
}

// out16[b][c][r] = (f16) in[b][r][c]
__global__ void transpose_cast_f16(const float* __restrict__ in, _Float16* __restrict__ out,
                                   int R, int Cc, long sIn, long sOut) {
  int b = blockIdx.z;
  in  += (size_t)b * sIn;
  out += (size_t)b * sOut;
  int idx = blockIdx.x * blockDim.x + threadIdx.x;
  if (idx >= R * Cc) return;
  int r = idx / Cc, c = idx % Cc;
  out[(size_t)c * R + r] = (_Float16)in[idx];
}

// cfv = (1+L5)*vv + L6 ; cfa = (1+L7)*aa + L8   (f16 outputs for the bilinear)
__global__ void combine_feats(const float* __restrict__ L5, const float* __restrict__ L6,
                              const float* __restrict__ L7, const float* __restrict__ L8,
                              const float* __restrict__ vv, const float* __restrict__ aa,
                              _Float16* __restrict__ cfv, _Float16* __restrict__ cfa, long n) {
  long i = (long)blockIdx.x * blockDim.x + threadIdx.x;
  if (i >= n) return;
  cfv[i] = (_Float16)((1.0f + L5[i]) * vv[i] + L6[i]);
  cfa[i] = (_Float16)((1.0f + L7[i]) * aa[i] + L8[i]);
}

__global__ void final_gate(const float* __restrict__ am, const float* __restrict__ img,
                           const float* __restrict__ aud, const float* __restrict__ t_o,
                           float* __restrict__ out1, float* __restrict__ out2, long n) {
  long i = (long)blockIdx.x * blockDim.x + threadIdx.x;
  if (i >= n) return;
  float t = t_o[0];
  float j = 1.0f / (1.0f + __expf(-am[i]));
  float Z = t * j * img[i] + (1.0f - j) * aud[i];
  out1[i] = Z + img[i];
  out2[i] = Z + aud[i];
}

// ---------------------------------------------------------------------------
extern "C" void kernel_launch(void* const* d_in, const int* in_sizes, int n_in,
                              void* d_out, int out_size, void* d_ws, size_t ws_size,
                              hipStream_t stream) {
  (void)in_sizes; (void)n_in; (void)out_size; (void)ws_size;
  constexpr int B = 4, S = 512, D = 256;
  constexpr int M = B * S;                 // 2048 tokens
  constexpr long ND = (long)M * D;         // 524288
  constexpr long NSS = (long)B * S * S;    // 1048576

  const float* img = (const float*)d_in[0];
  const float* aud = (const float*)d_in[1];
  const float* com = (const float*)d_in[2];
  const float* w[9], *bb[9];
  for (int i = 0; i < 9; ++i) { w[i] = (const float*)d_in[3 + 2 * i]; bb[i] = (const float*)d_in[4 + 2 * i]; }
  const float* bil = (const float*)d_in[21];
  const float* t_o = (const float*)d_in[22];

  // ---- workspace carve-up (256B aligned) ----
  char* Wp = (char*)d_ws;
  size_t off = 0;
  auto take = [&](size_t bytes) { void* p = Wp + off; off += (bytes + 255) & ~(size_t)255; return p; };

  _Float16* bil16 = (_Float16*)take((size_t)D * D * D * 2);   // 33.5 MB, L2-resident
  _Float16* img16 = (_Float16*)take(ND * 2);                  // row-major f16 inputs
  _Float16* aud16 = (_Float16*)take(ND * 2);
  _Float16* com16 = (_Float16*)take(ND * 2);
  _Float16* imgT  = (_Float16*)take(ND * 2);                  // per-batch transposed values
  _Float16* audT  = (_Float16*)take(ND * 2);
  _Float16* comT  = (_Float16*)take(ND * 2);
  _Float16* w16t[9];
  for (int i = 0; i < 9; ++i) w16t[i] = (_Float16*)take((size_t)D * D * 2);
  _Float16* qv16 = (_Float16*)take(ND * 2);
  _Float16* qa16 = (_Float16*)take(ND * 2);
  _Float16* kv16 = (_Float16*)take(ND * 2);
  _Float16* ka16 = (_Float16*)take(ND * 2);
  _Float16* cc16 = (_Float16*)take(ND * 2);
  float* sc_vv = (float*)take(NSS * 4);                       // pre-softmax scores (f32)
  float* sc_vc = (float*)take(NSS * 4);
  float* sc_aa = (float*)take(NSS * 4);
  float* sc_ac = (float*)take(NSS * 4);
  _Float16* at_vv = (_Float16*)take(NSS * 2);                 // post-softmax attn (f16)
  _Float16* at_vc = (_Float16*)take(NSS * 2);
  _Float16* at_aa = (_Float16*)take(NSS * 2);
  _Float16* at_ac = (_Float16*)take(NSS * 2);
  _Float16* vc16 = (_Float16*)take(ND * 2);                   // attn@common (A of w5..w8)
  _Float16* ac16 = (_Float16*)take(ND * 2);
  float* vv = (float*)take(ND * 4);                           // attn@img / attn@audio (f32)
  float* aa = (float*)take(ND * 4);
  float* L5 = (float*)take(ND * 4);
  float* L6 = (float*)take(ND * 4);
  float* L7 = (float*)take(ND * 4);
  float* L8 = (float*)take(ND * 4);
  _Float16* cfv16 = (_Float16*)take(ND * 2);
  _Float16* cfa16 = (_Float16*)take(ND * 2);
  float* am = (float*)take(ND * 4);

  // ---- Phase A: casts / transposes ----
  {
    long n = (long)D * D * D;
    cast_f16_kernel<<<dim3((n + 255) / 256), dim3(256), 0, stream>>>(bil, bil16, n);
  }
  {
    dim3 g((ND + 255) / 256), t(256);
    cast_f16_kernel<<<g, t, 0, stream>>>(img, img16, ND);
    cast_f16_kernel<<<g, t, 0, stream>>>(aud, aud16, ND);
    cast_f16_kernel<<<g, t, 0, stream>>>(com, com16, ND);
  }
  {
    dim3 g((S * D) / 256, 1, B), t(256);
    transpose_cast_f16<<<g, t, 0, stream>>>(img, imgT, S, D, (long)S * D, (long)S * D);
    transpose_cast_f16<<<g, t, 0, stream>>>(aud, audT, S, D, (long)S * D, (long)S * D);
    transpose_cast_f16<<<g, t, 0, stream>>>(com, comT, S, D, (long)S * D, (long)S * D);
  }
  for (int i = 0; i < 9; ++i)
    transpose_cast_f16<<<dim3((D * D) / 256), dim3(256), 0, stream>>>(w[i], w16t[i], D, D, 0, 0);

  // ---- Phase B: input linears (M=2048, N=K=256) -> f16 activations ----
  {
    dim3 g(D / 64, M / 64, 1), t(128);
    gemm_wmma_f16<<<g, t, 0, stream>>>(img16, w16t[0], bb[0], nullptr, qv16, M, D, D, 0, 0, 0);
    gemm_wmma_f16<<<g, t, 0, stream>>>(aud16, w16t[1], bb[1], nullptr, qa16, M, D, D, 0, 0, 0);
    gemm_wmma_f16<<<g, t, 0, stream>>>(img16, w16t[2], bb[2], nullptr, kv16, M, D, D, 0, 0, 0);
    gemm_wmma_f16<<<g, t, 0, stream>>>(aud16, w16t[3], bb[3], nullptr, ka16, M, D, D, 0, 0, 0);
    gemm_wmma_f16<<<g, t, 0, stream>>>(com16, w16t[4], bb[4], nullptr, cc16, M, D, D, 0, 0, 0);
  }

  // ---- Phase C: scores = q @ k^T  (batched: M=N=512, K=256) -> f32 ----
  {
    dim3 g(S / 64, S / 64, B), t(128);
    long sa = (long)S * D, sb = (long)S * D, sc = (long)S * S;
    gemm_wmma_f16<<<g, t, 0, stream>>>(qv16, kv16, nullptr, sc_vv, nullptr, S, S, D, sa, sb, sc);
    gemm_wmma_f16<<<g, t, 0, stream>>>(qv16, cc16, nullptr, sc_vc, nullptr, S, S, D, sa, sb, sc);
    gemm_wmma_f16<<<g, t, 0, stream>>>(qa16, ka16, nullptr, sc_aa, nullptr, S, S, D, sa, sb, sc);
    gemm_wmma_f16<<<g, t, 0, stream>>>(qa16, cc16, nullptr, sc_ac, nullptr, S, S, D, sa, sb, sc);
  }

  // ---- Phase D: softmax rows, * 1/sqrt(D) (reference order) -> f16 ----
  {
    const float norm = 0.0625f;   // 1/sqrt(256)
    dim3 g((B * S) / 8), t(256);
    softmax_scale_f16<<<g, t, 0, stream>>>(sc_vv, at_vv, S, norm);
    softmax_scale_f16<<<g, t, 0, stream>>>(sc_vc, at_vc, S, norm);
    softmax_scale_f16<<<g, t, 0, stream>>>(sc_aa, at_aa, S, norm);
    softmax_scale_f16<<<g, t, 0, stream>>>(sc_ac, at_ac, S, norm);
  }

  // ---- Phase E: attn @ V  (batched: M=512, N=256, K=512) ----
  {
    dim3 g(D / 64, S / 64, B), t(128);
    long sa = (long)S * S, sb = (long)D * S, sc = (long)S * D;
    gemm_wmma_f16<<<g, t, 0, stream>>>(at_vc, comT, nullptr, nullptr, vc16, S, D, S, sa, sb, sc);
    gemm_wmma_f16<<<g, t, 0, stream>>>(at_ac, comT, nullptr, nullptr, ac16, S, D, S, sa, sb, sc);
    gemm_wmma_f16<<<g, t, 0, stream>>>(at_vv, imgT, nullptr, vv, nullptr, S, D, S, sa, sb, sc);
    gemm_wmma_f16<<<g, t, 0, stream>>>(at_aa, audT, nullptr, aa, nullptr, S, D, S, sa, sb, sc);
  }

  // ---- Phase F: gating linears w5..w8 -> f32 ----
  {
    dim3 g(D / 64, M / 64, 1), t(128);
    gemm_wmma_f16<<<g, t, 0, stream>>>(vc16, w16t[5], bb[5], L5, nullptr, M, D, D, 0, 0, 0);
    gemm_wmma_f16<<<g, t, 0, stream>>>(vc16, w16t[6], bb[6], L6, nullptr, M, D, D, 0, 0, 0);
    gemm_wmma_f16<<<g, t, 0, stream>>>(ac16, w16t[7], bb[7], L7, nullptr, M, D, D, 0, 0, 0);
    gemm_wmma_f16<<<g, t, 0, stream>>>(ac16, w16t[8], bb[8], L8, nullptr, M, D, D, 0, 0, 0);
  }

  // ---- Phase G: common_feat_v / common_feat_a (f16 for bilinear) ----
  combine_feats<<<dim3((ND + 255) / 256), dim3(256), 0, stream>>>(L5, L6, L7, L8, vv, aa,
                                                                  cfv16, cfa16, ND);

  // ---- Phase H: bilinear (dominant: 2048 x 65536 x 256 implicit GEMM) ----
  {
    dim3 g(D / 64, M / 128, 1), t(128);
    bilinear_wmma<<<g, t, 0, stream>>>(cfv16, cfa16, bil16, am, D);
  }

  // ---- Phase I: sigmoid gate + residual outputs ----
  {
    float* out1 = (float*)d_out;
    float* out2 = out1 + ND;
    final_gate<<<dim3((ND + 255) / 256), dim3(256), 0, stream>>>(am, img, aud, t_o, out1, out2, ND);
  }
}
